// EmbedNet_87136296501547
// MI455X (gfx1250) — matrix operand order repaired and checked by reading, not compiled, not run on hardware
//
#include <hip/hip_runtime.h>
#include <hip/hip_bf16.h>
#include <cmath>

typedef _Float16 f16;
typedef _Float16 v16h __attribute__((ext_vector_type(16)));
typedef float    v8f  __attribute__((ext_vector_type(8)));

#define BATCH 1024
#define NNODE 10
#define EPAD  112              // padded edges per batch = 7 WMMA row-tiles of 16
#define NEDGE (BATCH*EPAD)
#define SILU_GAIN 1.679177f

// ---------------- workspace layout (bytes) ----------------
#define OFF_CNT 0u            // int mask counter
#define OFF_CG  64u           // 783 f32: real CG tables, 15 combos
#define OFF_A   4096u         // 10240 x 16 f32
#define OFF_W2T 659456u       // 64x64 f16 transposed, 1/8 folded
#define OFF_W3T 667648u
#define OFF_W4T 675840u       // 624x64 f16 transposed, 1/8 folded
#define OFF_H1  755712u       // NEDGE x 64 f16
#define OFF_S   15435776u     // NEDGE x 48 f32 (43 used)
#define OFF_J   37455872u     // 1024 x 10 x 432 f32

struct KP {
  const float *R;
  const float *oh1_w1,*oh1_b1,*oh1_w2,*oh1_b2;
  const float *oh2_w1,*oh2_b1,*oh2_w2,*oh2_b2;
  const float *fit1_w1,*fit1_b1,*fit1_w2,*fit1_b2;
  const float *fit2_w1,*fit2_b1,*fit2_w2,*fit2_b2;
  const float *mlp_w1,*mlp_b1,*mlp_w2,*mlp_b2;
  const float *tp3_w,*fc0,*fc1,*fc2,*fc3,*lin_w;
  int   *cnt;
  float *cg;
  float *A;
  f16   *W2t,*W3t,*W4t,*H1;
  float *S,*J;
  float *out;
};

// ---------------- CG combo registry ----------------
__constant__ int CG_L[15][3] = {
  {0,0,0},{0,1,1},{0,2,2},{1,0,1},{1,1,0},{1,1,2},{1,2,1},{1,2,3},
  {2,0,2},{2,1,1},{2,1,3},{2,2,0},{2,2,2},{3,1,2},{3,2,1}};
__constant__ int CG_OFF[15] = {0,1,10,35,44,53,98,143,248,273,318,423,448,573,678};

// node TP (G x Y -> A), make_paths order, weight index = p
__constant__ int NPL[13][3] = {
  {0,0,0},{0,1,1},{0,2,2},{1,0,1},{1,1,0},{1,1,2},{1,2,1},{1,2,3},
  {2,0,2},{2,1,1},{2,1,3},{2,2,0},{2,2,2}};
// conv TP (A x sh -> msg), make_paths order; path p owns radial cols [48p,48p+48)
__constant__ int CPL[13][3] = {
  {0,0,0},{0,1,1},{0,2,2},{1,0,1},{1,1,0},{1,1,2},{1,2,1},
  {2,0,2},{2,1,1},{2,2,0},{2,2,2},{3,1,2},{3,2,1}};
__constant__ int CP_SOFF[13] = {0,1,4,9,12,13,18,21,26,29,30,35,40}; // 43 total
__constant__ int CP_OB[13]   = {0,48,192,48,0,192,48,192,48,0,192,192,48};
__constant__ int IOFF4[4] = {0,1,4,9};   // irrep offsets within 16 (l=0..3)
__constant__ int FANN[4]  = {3,4,4,2};   // node TP fan by l3
__constant__ int FANC[3]  = {3,5,5};     // conv TP fan by l3

__device__ __forceinline__ int cgoff(int l1,int l2,int l3){
  #pragma unroll
  for (int i=0;i<15;++i)
    if (CG_L[i][0]==l1 && CG_L[i][1]==l2 && CG_L[i][2]==l3) return CG_OFF[i];
  return 0;
}
__device__ __forceinline__ float siluf(float x){ return x / (1.f + __expf(-x)); }

// ---------------- k0: build exact real-CG tables + weight prep ----------------
__device__ double dfact(int n){ double r=1.0; for(int i=2;i<=n;++i) r*=(double)i; return r; }

__device__ double cgc_dev(int j1,int j2,int j3,int m1,int m2,int m3){
  if (m1+m2 != m3) return 0.0;
  double pre = sqrt((double)(2*j3+1)*dfact(j3+j1-j2)*dfact(j3-j1+j2)*dfact(j1+j2-j3)/dfact(j1+j2+j3+1));
  pre *= sqrt(dfact(j3+m3)*dfact(j3-m3)*dfact(j1-m1)*dfact(j1+m1)*dfact(j2-m2)*dfact(j2+m2));
  double s = 0.0;
  for (int k=0;k<=j1+j2+j3;++k){
    int d0=k, d1=j1+j2-j3-k, d2=j1-m1-k, d3=j2+m2-k, d4=j3-j2+m1+k, d5=j3-j1-m2+k;
    if (d0<0||d1<0||d2<0||d3<0||d4<0||d5<0) continue;
    double dn = dfact(d0)*dfact(d1)*dfact(d2)*dfact(d3)*dfact(d4)*dfact(d5);
    s += ((k&1)? -1.0:1.0)/dn;
  }
  return pre*s;
}
__device__ void fillU(int l, double Ur[7][7], double Ui[7][7]){
  for (int i=0;i<7;++i) for (int j=0;j<7;++j){ Ur[i][j]=0; Ui[i][j]=0; }
  double s2 = sqrt(0.5);
  for (int mr=-l; mr<=l; ++mr){
    int r = mr+l;
    if (mr<0){
      Ui[r][l+mr] = s2;
      double sg = ((-mr)&1)? -1.0:1.0;
      Ui[r][l-mr] = -sg*s2;
    } else if (mr==0){ Ur[r][l] = 1.0; }
    else {
      Ur[r][l-mr] = s2;
      double sg = (mr&1)? -1.0:1.0;
      Ur[r][l+mr] = sg*s2;
    }
  }
}
__device__ void buildCG(int ci, float* cgbase){
  int l1=CG_L[ci][0], l2=CG_L[ci][1], l3=CG_L[ci][2];
  int d1=2*l1+1, d2=2*l2+1, d3=2*l3+1;
  double Cc[7][7][7];
  for (int a=0;a<7;++a) for (int b=0;b<7;++b) for (int c=0;c<7;++c) Cc[a][b][c]=0.0;
  for (int m1=-l1;m1<=l1;++m1) for (int m2=-l2;m2<=l2;++m2){
    int m3=m1+m2;
    if (m3>=-l3 && m3<=l3) Cc[m1+l1][m2+l2][m3+l3] = cgc_dev(l1,l2,l3,m1,m2,m3);
  }
  double U1r[7][7],U1i[7][7],U2r[7][7],U2i[7][7],U3r[7][7],U3i[7][7];
  fillU(l1,U1r,U1i); fillU(l2,U2r,U2i); fillU(l3,U3r,U3i);
  double Cr[7][7][7], Cim[7][7][7];
  double best=-1.0, zr=1.0, zi=0.0;
  for (int a=0;a<d1;++a) for (int b=0;b<d2;++b) for (int c=0;c<d3;++c){
    double sr=0.0, si=0.0;
    for (int m=0;m<d1;++m){
      double u1r=U1r[a][m], u1i=U1i[a][m];
      if (u1r==0.0 && u1i==0.0) continue;
      for (int n=0;n<d2;++n){
        double u2r=U2r[b][n], u2i=U2i[b][n];
        if (u2r==0.0 && u2i==0.0) continue;
        double tr = u1r*u2r - u1i*u2i;
        double ti = u1r*u2i + u1i*u2r;
        for (int k=0;k<d3;++k){
          double cc = Cc[m][n][k];
          if (cc==0.0) continue;
          double u3r=U3r[c][k], u3i=-U3i[c][k];    // conj
          sr += (tr*u3r - ti*u3i)*cc;
          si += (tr*u3i + ti*u3r)*cc;
        }
      }
    }
    Cr[a][b][c]=sr; Cim[a][b][c]=si;
    double mag = sr*sr + si*si;
    if (mag > best){ best=mag; zr=sr; zi=si; }
  }
  double zn = sqrt(best);
  float* dst = cgbase + CG_OFF[ci];
  for (int a=0;a<d1;++a) for (int b=0;b<d2;++b) for (int c=0;c<d3;++c)
    dst[(a*d2+b)*d3+c] = (float)((Cr[a][b][c]*zr + Cim[a][b][c]*zi)/zn);
}

__global__ void k0_init(KP P){
  int tid = threadIdx.x;
  if (tid == 0) *P.cnt = 0;
  if (tid < 15) buildCG(tid, P.cg);
  // f16 transposed weights with 1/sqrt(64) folded
  for (int q = tid; q < 64*64; q += 256){
    int nn = q >> 6, kk = q & 63;
    P.W2t[q] = (f16)(P.fc1[kk*64+nn]*0.125f);
    P.W3t[q] = (f16)(P.fc2[kk*64+nn]*0.125f);
  }
  for (int q = tid; q < 624*64; q += 256){
    int nn = q >> 6, kk = q & 63;
    P.W4t[q] = (f16)(P.fc3[kk*624+nn]*0.125f);
  }
}

// ---------------- k1: per-node MLPs, G/Y spherical harmonics, A = fctp(G,Y) ----------------
__device__ void shComp(float vx,float vy,float vz,float* o){
  float sq = vx*vx+vy*vy+vz*vz;
  float inv = sq>0.f ? rsqrtf(sq) : 0.f;
  float x=vx*inv, y=vy*inv, z=vz*inv;
  const float r3=1.7320508f, r15=3.8729833f, r5h=1.1180340f, r15h=1.9364917f;
  o[0]=1.f; o[1]=r3*y; o[2]=r3*z; o[3]=r3*x;
  o[4]=r15*x*y; o[5]=r15*y*z; o[6]=r5h*(3.f*z*z-1.f); o[7]=r15*x*z; o[8]=r15h*(x*x-y*y);
}

__global__ void k1_node(KP P){
  int idx = blockIdx.x*blockDim.x + threadIdx.x;
  if (idx >= BATCH*NNODE) return;
  const float* r = P.R + (size_t)idx*6;
  int i5 = (int)r[4]; i5 = i5<0?0:(i5>9?9:i5);
  int i6 = (int)r[5]; i6 = i6<0?0:(i6>9?9:i6);
  float O, Bv;
  { float acc = P.oh1_b2[0];
    for (int j=0;j<32;++j) acc += siluf(P.oh1_w1[i5*32+j]+P.oh1_b1[j])*P.oh1_w2[j];
    float a2 = P.fit1_b2[0];
    for (int j=0;j<4;++j)  a2 += siluf(acc*P.fit1_w1[j]+P.fit1_b1[j])*P.fit1_w2[j];
    O = a2; }
  { float acc = P.oh2_b2[0];
    for (int j=0;j<32;++j) acc += siluf(P.oh2_w1[i6*32+j]+P.oh2_b1[j])*P.oh2_w2[j];
    float a2 = P.fit2_b2[0];
    for (int j=0;j<4;++j)  a2 += siluf(acc*P.fit2_w1[j]+P.fit2_b1[j])*P.fit2_w2[j];
    Bv = a2; }
  float g0=r[0], g1=O, g2=Bv;
  float ov[3] = {P.mlp_b2[0], P.mlp_b2[1], P.mlp_b2[2]};
  for (int j=0;j<32;++j){
    float h = siluf(g0*P.mlp_w1[j] + g1*P.mlp_w1[32+j] + g2*P.mlp_w1[64+j] + P.mlp_b1[j]);
    ov[0]+=h*P.mlp_w2[j*3+0]; ov[1]+=h*P.mlp_w2[j*3+1]; ov[2]+=h*P.mlp_w2[j*3+2];
  }
  float G[9], Y[9];
  shComp(ov[0],ov[1],ov[2],G);
  shComp(r[1],r[2],r[3],Y);
  float A16[16];
  for (int q=0;q<16;++q) A16[q]=0.f;
  for (int p=0;p<13;++p){
    int l1=NPL[p][0], l2=NPL[p][1], l3=NPL[p][2];
    const float* cg = P.cg + cgoff(l1,l2,l3);
    int d1=2*l1+1, d2=2*l2+1, d3=2*l3+1;
    int o1=IOFF4[l1], o2=IOFF4[l2], o3=IOFF4[l3];
    float sc = rsqrtf((float)(FANN[l3]*(2*l3+1))) * P.tp3_w[p];
    for (int k=0;k<d3;++k){
      float acc=0.f;
      for (int ii=0;ii<d1;++ii)
        for (int jj=0;jj<d2;++jj)
          acc += cg[(ii*d2+jj)*d3+k]*G[o1+ii]*Y[o2+jj];
      A16[o3+k] += sc*acc;
    }
  }
  float* Ao = P.A + (size_t)idx*16;
  for (int q=0;q<16;++q) Ao[q]=A16[q];
}

// ---------------- k2: per-edge sh, radial embedding, layer1, CG values s ----------------
__global__ void k2_edge(KP P){
  int idx = blockIdx.x*blockDim.x + threadIdx.x;
  if (idx >= NEDGE) return;
  int b = idx / EPAD, e = idx % EPAD;
  f16*   h1row = P.H1 + (size_t)idx*64;
  float* srow  = P.S  + (size_t)idx*48;
  if (e >= 100){
    for (int q=0;q<64;++q) h1row[q]=(f16)0.f;
    for (int q=0;q<48;++q) srow[q]=0.f;
    return;
  }
  int i = e/10, j = e%10;
  const float* Rb = P.R + (size_t)(b*NNODE)*6;
  float px = Rb[j*6+1]-Rb[i*6+1];
  float py = Rb[j*6+2]-Rb[i*6+2];
  float pz = Rb[j*6+3]-Rb[i*6+3];
  float sq = px*px+py*py+pz*pz;
  float d  = sq>0.f ? sqrtf(sq) : 0.f;
  bool mask = (d < 8.0f) && (i != j);
  if (mask) atomicAdd(P.cnt, 1);
  // sh with 'norm' normalization
  float sh[9];
  { float inv = sq>0.f ? rsqrtf(sq) : 0.f;
    float x=px*inv, y=py*inv, z=pz*inv;
    const float r3=1.7320508f;
    sh[0]=1.f; sh[1]=y; sh[2]=z; sh[3]=x;
    sh[4]=r3*x*y; sh[5]=r3*y*z; sh[6]=0.5f*(3.f*z*z-1.f); sh[7]=r3*x*z; sh[8]=0.5f*r3*(x*x-y*y); }
  // soft_one_hot * sqrt(4)
  float emb[4];
  #pragma unroll
  for (int k=0;k<4;++k){
    float c = 1.6f*(float)(k+1);
    float dd = (d-c)*0.625f;
    emb[k] = __expf(-dd*dd)*(2.0f/1.12f);
  }
  // radial layer 1: 4 -> 64, weight/2, silu * gain, store f16 row-major
  for (int col=0;col<64;++col){
    float acc = emb[0]*P.fc0[col] + emb[1]*P.fc0[64+col] + emb[2]*P.fc0[128+col] + emb[3]*P.fc0[192+col];
    acc *= 0.5f;
    acc = siluf(acc)*SILU_GAIN;
    h1row[col] = (f16)acc;
  }
  // s_{p,k} = sc_p * mask * sum_ij cg[i,j,k] * A_i[..] * sh[..]
  const float* Arow = P.A + (size_t)(b*NNODE+i)*16;
  float mf = mask ? 1.f : 0.f;
  for (int p=0;p<13;++p){
    int l1=CPL[p][0], l2=CPL[p][1], l3=CPL[p][2];
    const float* cg = P.cg + cgoff(l1,l2,l3);
    int d1=2*l1+1, d2=2*l2+1, d3=2*l3+1;
    int o1=IOFF4[l1], o2=IOFF4[l2];
    float sc = rsqrtf((float)FANC[l3]) * mf;
    for (int k=0;k<d3;++k){
      float acc=0.f;
      for (int ii=0;ii<d1;++ii)
        for (int jj=0;jj<d2;++jj)
          acc += cg[(ii*d2+jj)*d3+k]*Arow[o1+ii]*sh[o2+jj];
      srow[CP_SOFF[p]+k] = acc*sc;
    }
  }
  srow[43]=0.f; srow[44]=0.f; srow[45]=0.f; srow[46]=0.f; srow[47]=0.f;
}

// ---------------- k3: WMMA radial MLP + fused CG contraction ----------------
__device__ __forceinline__ v16h load_a_frag(const f16* hrow, int chunk, int hs){
  // 16-bit A 16x32 layout: lanes<16 v0-3:K=0..7 v4-7:K=16..23; lanes>=16 shifted by 8
  v16h a;
  const f16* p0 = hrow + chunk*32 + 8*hs;
  const f16* p1 = p0 + 16;
  #pragma unroll
  for (int q=0;q<8;++q){ a[q]=p0[q]; a[8+q]=p1[q]; }
  return a;
}
__device__ __forceinline__ v16h load_b_frag(const f16* wrow, int chunk, int hs){
  // B 32x16: lane holds one column; lanes<16 v0-3:K=0..7 v4-7:K=16..23; lanes>=16 shifted by 8
  v16h bf;
  const f16* p0 = wrow + chunk*32 + 8*hs;
  const f16* p1 = p0 + 16;
  #pragma unroll
  for (int q=0;q<8;++q){ bf[q]=p0[q]; bf[8+q]=p1[q]; }
  return bf;
}
__device__ __forceinline__ v8f wmma_f16(v16h a, v16h b, v8f c){
  return __builtin_amdgcn_wmma_f32_16x16x32_f16(false, a, false, b, (short)0, c, false, false);
}

__global__ void __launch_bounds__(224)
k3_conv(KP P){
  __shared__ __align__(16) f16   lds_h[7][16][64];
  __shared__ __align__(16) float lds_s[7][16][48];
  __shared__ float lds_out[NNODE][432];
  int lane = threadIdx.x & 31;
  int wv   = threadIdx.x >> 5;          // 0..6 : edge row-tile
  int hs   = lane >> 4;
  int ln   = lane & 15;
  size_t tileBase = (size_t)blockIdx.x*EPAD + wv*16;

  // stage h1 tile (16x64 f16) and s tile (16x48 f32) into LDS; zero out-accumulator
  { const uint4* src = (const uint4*)(P.H1 + tileBase*64);
    uint4* dst = (uint4*)&lds_h[wv][0][0];
    for (int q=lane;q<128;q+=32) dst[q]=src[q];
    const uint4* ss = (const uint4*)(P.S + tileBase*48);
    uint4* sd = (uint4*)&lds_s[wv][0][0];
    for (int q=lane;q<192;q+=32) sd[q]=ss[q]; }
  for (int q=threadIdx.x; q<NNODE*432; q+=blockDim.x) (&lds_out[0][0])[q]=0.f;
  __syncthreads();

  // hidden layers 64->64 (x2): 4 N-tiles x 2 K-chunks of v_wmma_f32_16x16x32_f16
  const f16* Wts0 = P.W2t;
  const f16* Wts1 = P.W3t;
  for (int layer=0; layer<2; ++layer){
    const f16* Wt = layer ? Wts1 : Wts0;
    const f16* hrow = &lds_h[wv][ln][0];
    v16h a0 = load_a_frag(hrow, 0, hs);
    v16h a1 = load_a_frag(hrow, 1, hs);
    v8f accv[4];
    #pragma unroll
    for (int nt=0; nt<4; ++nt){
      int ncol = nt*16 + ln;
      v16h b0 = load_b_frag(Wt + ncol*64, 0, hs);
      v16h b1 = load_b_frag(Wt + ncol*64, 1, hs);
      v8f c = {0.f,0.f,0.f,0.f,0.f,0.f,0.f,0.f};
      c = wmma_f16(a0,b0,c);
      c = wmma_f16(a1,b1,c);
      accv[nt]=c;
    }
    // silu*gain, restage f16 (all reads above precede stores; per-wave slab)
    #pragma unroll
    for (int nt=0; nt<4; ++nt){
      int col = nt*16 + ln;
      #pragma unroll
      for (int r=0;r<8;++r){
        float x = accv[nt][r];
        x = SILU_GAIN * (x / (1.f + __expf(-x)));
        lds_h[wv][r+8*hs][col] = (f16)x;
      }
    }
    __syncthreads();
  }

  // layer 4 (64->624) fused with CG contraction + scatter into per-batch accumulator
  const f16* hrow = &lds_h[wv][ln][0];
  v16h a0 = load_a_frag(hrow, 0, hs);
  v16h a1 = load_a_frag(hrow, 1, hs);
  int jr[8];
  #pragma unroll
  for (int r=0;r<8;++r){ int e = wv*16 + r + 8*hs; jr[r] = e - (e/10)*10; }
  for (int t=0; t<39; ++t){
    int p = t/3;
    int ncol = t*16 + ln;                 // 0..623
    const f16* wrow = P.W4t + ncol*64;
    if (t < 38) __builtin_prefetch(P.W4t + (ncol+16)*64, 0, 1);
    v16h b0 = load_b_frag(wrow, 0, hs);
    v16h b1 = load_b_frag(wrow, 1, hs);
    v8f c = {0.f,0.f,0.f,0.f,0.f,0.f,0.f,0.f};
    c = wmma_f16(a0,b0,c);
    c = wmma_f16(a1,b1,c);
    int l3 = CPL[p][2];
    int dp = 2*l3+1;
    int chan = ncol - p*48;               // 0..47
    int ob = CP_OB[p] + chan*dp;
    int so = CP_SOFF[p];
    #pragma unroll
    for (int r=0;r<8;++r){
      float dv = c[r];
      int m = r + 8*hs;
      const float* sr = &lds_s[wv][m][so];
      float* orow = &lds_out[jr[r]][ob];
      for (int k=0;k<dp;++k) atomicAdd(&orow[k], dv*sr[k]);
    }
  }
  __syncthreads();
  float* Jb = P.J + (size_t)blockIdx.x*(NNODE*432);
  for (int q=threadIdx.x; q<NNODE*432; q+=blockDim.x) Jb[q] = (&lds_out[0][0])[q];
}

// ---------------- k4: o3_linear432 + degree normalization ----------------
__global__ void k4_linear(KP P){
  int idx = blockIdx.x*blockDim.x + threadIdx.x;
  if (idx >= BATCH*NNODE*432) return;
  int node = idx / 432, r = idx % 432;
  int l, d, base;
  if (r < 48)       { l=0; d=1; base=0;   }
  else if (r < 192) { l=1; d=3; base=48;  }
  else              { l=2; d=5; base=192; }
  int v = (r-base)/d, i = (r-base)%d;
  const float* Jrow = P.J + (size_t)node*432 + base + i;
  const float* W = P.lin_w + l*48*48;
  float acc = 0.f;
  for (int u=0;u<48;++u) acc += Jrow[u*d]*W[u*48+v];
  float avg_deg = (float)(*P.cnt) / (float)(BATCH*NNODE);
  float scale = rsqrtf(avg_deg) * 0.14433757f;   // 1/sqrt(48)
  P.out[idx] = acc*scale;
}

// ---------------- launch ----------------
extern "C" void kernel_launch(void* const* d_in, const int* in_sizes, int n_in,
                              void* d_out, int out_size, void* d_ws, size_t ws_size,
                              hipStream_t stream){
  (void)in_sizes; (void)n_in; (void)out_size; (void)ws_size;
  char* w = (char*)d_ws;
  KP P;
  // setup_inputs() insertion order: R, then params dict (insertion order),
  // fc_ws list expanded in order.
  P.R       = (const float*)d_in[0];
  P.oh1_w1  = (const float*)d_in[1];  P.oh1_b1 = (const float*)d_in[2];
  P.oh1_w2  = (const float*)d_in[3];  P.oh1_b2 = (const float*)d_in[4];
  P.oh2_w1  = (const float*)d_in[5];  P.oh2_b1 = (const float*)d_in[6];
  P.oh2_w2  = (const float*)d_in[7];  P.oh2_b2 = (const float*)d_in[8];
  P.fit1_w1 = (const float*)d_in[9];  P.fit1_b1= (const float*)d_in[10];
  P.fit1_w2 = (const float*)d_in[11]; P.fit1_b2= (const float*)d_in[12];
  P.fit2_w1 = (const float*)d_in[13]; P.fit2_b1= (const float*)d_in[14];
  P.fit2_w2 = (const float*)d_in[15]; P.fit2_b2= (const float*)d_in[16];
  P.mlp_w1  = (const float*)d_in[17]; P.mlp_b1 = (const float*)d_in[18];
  P.mlp_w2  = (const float*)d_in[19]; P.mlp_b2 = (const float*)d_in[20];
  P.tp3_w   = (const float*)d_in[21];
  P.fc0     = (const float*)d_in[22]; P.fc1 = (const float*)d_in[23];
  P.fc2     = (const float*)d_in[24]; P.fc3 = (const float*)d_in[25];
  P.lin_w   = (const float*)d_in[26];
  P.cnt = (int*)  (w + OFF_CNT);
  P.cg  = (float*)(w + OFF_CG);
  P.A   = (float*)(w + OFF_A);
  P.W2t = (f16*)  (w + OFF_W2T);
  P.W3t = (f16*)  (w + OFF_W3T);
  P.W4t = (f16*)  (w + OFF_W4T);
  P.H1  = (f16*)  (w + OFF_H1);
  P.S   = (float*)(w + OFF_S);
  P.J   = (float*)(w + OFF_J);
  P.out = (float*)d_out;

  k0_init  <<<1,               256, 0, stream>>>(P);
  k1_node  <<<40,              256, 0, stream>>>(P);
  k2_edge  <<<NEDGE/256,       256, 0, stream>>>(P);
  k3_conv  <<<BATCH,           224, 0, stream>>>(P);
  k4_linear<<<(BATCH*NNODE*432)/256, 256, 0, stream>>>(P);
}
